// SSAAA_88252987998909
// MI455X (gfx1250) — matrix operand order repaired and checked by reading, not compiled, not run on hardware
//
#include <hip/hip_runtime.h>
#include <hip/hip_bf16.h>
#include <stdint.h>

// ---------------------------------------------------------------------------
// Types for CDNA5 WMMA (wave32): 16x16x32 bf16 -> f32
// ---------------------------------------------------------------------------
typedef __attribute__((ext_vector_type(16))) __bf16 v16bf;
typedef __attribute__((ext_vector_type(8)))  float  v8f;

static __device__ __forceinline__ unsigned short f2bf(float x) {
    union { float f; uint32_t u; } c; c.f = x;
    uint32_t r = c.u + 0x7FFFu + ((c.u >> 16) & 1u);   // round-to-nearest-even
    return (unsigned short)(r >> 16);
}

// ---------------------------------------------------------------------------
// cast f32 [R,C] (contiguous) -> bf16 [R,CP] zero-padded columns
// ---------------------------------------------------------------------------
__global__ void cast_pad_kernel(const float* __restrict__ in,
                                unsigned short* __restrict__ out,
                                int C, int CP, long total) {
    long p = (long)blockIdx.x * blockDim.x + threadIdx.x;
    if (p >= total) return;
    int c = (int)(p % CP);
    long r = p / CP;
    out[p] = (c < C) ? f2bf(in[r * C + c]) : (unsigned short)0;
}

// ---------------------------------------------------------------------------
// transpose + cast: in f32 [Rin x Cin] (ld=ldin)  ->  out bf16 [Rout x Cout]
//   out[o*Cout + i] = (o<Cin && i<Rin) ? in[i*ldin + o] : 0
// LDS-tiled so both global read and write are coalesced.
// ---------------------------------------------------------------------------
__global__ __launch_bounds__(256) void transpose_cast_kernel(
        const float* __restrict__ in, unsigned short* __restrict__ out,
        int Rin, int Cin, int ldin, int Rout, int Cout) {
    __shared__ float tile[16][17];
    int ibase = blockIdx.x * 16;     // i: column index of out (row of in)
    int obase = blockIdx.y * 16;     // o: row index of out (col of in)
    int tx = threadIdx.x & 15, ty = threadIdx.x >> 4;
    int r = ibase + ty, c = obase + tx;
    tile[ty][tx] = (r < Rin && c < Cin) ? in[(size_t)r * ldin + c] : 0.f;
    __syncthreads();
    int o = obase + ty, i = ibase + tx;
    if (o < Rout && i < Cout)
        out[(size_t)o * Cout + i] = f2bf(tile[tx][ty]);
}

// ---------------------------------------------------------------------------
// WMMA GEMM:  C[M,NP] (f32) = A[M,K] (bf16, row-major) x Bt[NP,K]^T (bf16)
// M % 32 == 0, K % 64 == 0, NP % 256 == 0.  Block = 128 threads = 4 waves.
// Each wave computes a 32x64 tile (2 M-tiles x 4 N-tiles = 8 WMMAs/half-step);
// a block covers 32x256.  Software pipelining is unrolled by 2 with two NAMED
// fragment buffer sets so every index is a compile-time constant (keeps the
// fragments in VGPRs — no scratch/LDS demotion) while the loads for one
// half-step remain in flight during the other half-step's WMMAs.
// Per-lane fragment chunks are contiguous 16B/32B per the ISA wave32 layout.
// ---------------------------------------------------------------------------
union Frag { v16bf v; uint4 q[2]; };

__global__ __launch_bounds__(128) void gemm_bf16_wmma_kernel(
        const unsigned short* __restrict__ A,
        const unsigned short* __restrict__ Bt,
        float* __restrict__ C, int K, int NP) {
    const int m0   = blockIdx.y << 5;                 // 32 rows per block
    const int wave = threadIdx.x >> 5;
    const int lane = threadIdx.x & 31;
    const int hl   = lane >> 4;                       // 0: lanes 0-15, 1: 16-31
    const int lm   = lane & 15;
    const int nb   = (blockIdx.x << 8) + (wave << 6); // wave's 64-col slab

    v8f acc[2][4];
#pragma unroll
    for (int mt = 0; mt < 2; ++mt)
#pragma unroll
        for (int t = 0; t < 4; ++t)
            acc[mt][t] = (v8f){0.f,0.f,0.f,0.f,0.f,0.f,0.f,0.f};

    // A fragment base: lane (hl,lm) reads A[m0+16*mt+lm, k0+8*hl .. +8) and
    // A[.., k0+16+8*hl .. +8)  (two 16B chunks).
    const unsigned short* arow0 = A + (size_t)(m0 + lm) * K + hl * 8;
    const unsigned short* arow1 = arow0 + (size_t)16 * K;
    // B fragment base: lane reads Bt[nb+16t+lm, k0+8*hl .. +8) and
    // Bt[.., k0+16+8*hl .. +8)  (two 16B chunks).
    const unsigned short* brow[4];
#pragma unroll
    for (int t = 0; t < 4; ++t)
        brow[t] = Bt + (size_t)(nb + t * 16 + lm) * K + hl * 8;

    Frag fa0[2], fb0[4], fa1[2], fb1[4];

    auto load_frags = [&](Frag (&fa)[2], Frag (&fb)[4], int k) {
        fa[0].q[0] = *(const uint4*)(arow0 + k);
        fa[0].q[1] = *(const uint4*)(arow0 + k + 16);
        fa[1].q[0] = *(const uint4*)(arow1 + k);
        fa[1].q[1] = *(const uint4*)(arow1 + k + 16);
#pragma unroll
        for (int t = 0; t < 4; ++t) {
            fb[t].q[0] = *(const uint4*)(brow[t] + k);
            fb[t].q[1] = *(const uint4*)(brow[t] + k + 16);
        }
    };
    auto do_mmas = [&](Frag (&fa)[2], Frag (&fb)[4]) {
#pragma unroll
        for (int t = 0; t < 4; ++t) {
            acc[0][t] = __builtin_amdgcn_wmma_f32_16x16x32_bf16(
                            false, fa[0].v, false, fb[t].v,
                            (short)0, acc[0][t], false, false);
            acc[1][t] = __builtin_amdgcn_wmma_f32_16x16x32_bf16(
                            false, fa[1].v, false, fb[t].v,
                            (short)0, acc[1][t], false, false);
        }
    };

    // Prologue: buffer 0 holds k = 0.
    load_frags(fa0, fb0, 0);

    for (int k0 = 0; k0 < K; k0 += 64) {
        // Issue loads for k0+32 (buffer 1), then compute on buffer 0.
        load_frags(fa1, fb1, k0 + 32);
        if (k0 + 64 < K) {
            __builtin_prefetch(arow0 + k0 + 64, 0, 1);    // global_prefetch_b8
            __builtin_prefetch(brow[0] + k0 + 64, 0, 1);
        }
        do_mmas(fa0, fb0);
        // Issue loads for k0+64 (buffer 0), then compute on buffer 1.
        if (k0 + 64 < K)
            load_frags(fa0, fb0, k0 + 64);
        do_mmas(fa1, fb1);
    }

    // D layout: VGPR r, lane l -> C[m_tile + r + 8*hl, nb + 16t + lm]
#pragma unroll
    for (int mt = 0; mt < 2; ++mt) {
#pragma unroll
        for (int t = 0; t < 4; ++t) {
            float* crow = C + (size_t)(m0 + mt * 16 + hl * 8) * NP + (nb + t * 16 + lm);
#pragma unroll
            for (int r = 0; r < 8; ++r)
                crow[(size_t)r * NP] = acc[mt][t][r];
        }
    }
}

// ---------------------------------------------------------------------------
// Per-row dot products: s1[i] = Wh[i,:].a[0:F], s2[i] = Wh[i,:].a[F:2F]
// ---------------------------------------------------------------------------
__global__ __launch_bounds__(256) void s1s2_kernel(
        const float* __restrict__ Wh, const float* __restrict__ a,
        float* __restrict__ s1, float* __restrict__ s2, int F, int ld) {
    __shared__ float r1[256], r2[256];
    int i = blockIdx.x, t = threadIdx.x;
    float p1 = 0.f, p2 = 0.f;
    for (int c = t; c < F; c += 256) {
        float w = Wh[(size_t)i * ld + c];
        p1 += w * a[c];
        p2 += w * a[F + c];
    }
    r1[t] = p1; r2[t] = p2; __syncthreads();
    for (int s = 128; s > 0; s >>= 1) {
        if (t < s) { r1[t] += r1[t + s]; r2[t] += r2[t + s]; }
        __syncthreads();
    }
    if (t == 0) { s1[i] = r1[0]; s2[i] = r2[0]; }
}

// ---------------------------------------------------------------------------
// Masked softmax row kernel (online max/sum, then one write pass).
// e_ij = adj>0 ? leaky_relu(s1_i + s2_j, 0.2) : -9e15 ; att = softmax_j(e)
// Output directly in bf16 for the following WMMA GEMM.
// ---------------------------------------------------------------------------
__global__ __launch_bounds__(256) void attn_softmax_kernel(
        const int* __restrict__ adj, const float* __restrict__ s1,
        const float* __restrict__ s2, unsigned short* __restrict__ att, int Nn) {
    __shared__ float sm[256], sl[256];
    int i = blockIdx.x, t = threadIdx.x;
    float s1i = s1[i];
    const int* arow = adj + (size_t)i * Nn;
    float m = -3.0e38f, l = 0.f;
    for (int j = t; j < Nn; j += 256) {
        float e = s1i + s2[j];
        e = (e > 0.f) ? e : 0.2f * e;
        if (arow[j] <= 0) e = -9.0e15f;
        if (e > m) { l *= __expf(m - e); m = e; }
        l += __expf(e - m);
    }
    sm[t] = m; sl[t] = l; __syncthreads();
    for (int s = 128; s > 0; s >>= 1) {
        if (t < s) {
            float m2 = sm[t + s], l2 = sl[t + s];
            float M = fmaxf(sm[t], m2);
            sl[t] = sl[t] * __expf(sm[t] - M) + l2 * __expf(m2 - M);
            sm[t] = M;
        }
        __syncthreads();
    }
    float M = sm[0];
    float inv = 1.0f / sl[0];
    for (int j = t; j < Nn; j += 256) {
        float e = s1i + s2[j];
        e = (e > 0.f) ? e : 0.2f * e;
        if (arow[j] <= 0) e = -9.0e15f;
        att[(size_t)i * Nn + j] = f2bf(__expf(e - M) * inv);
    }
}

// ---------------------------------------------------------------------------
// ELU: out[r*C+c] = elu(in[r*ldin+c])
// ---------------------------------------------------------------------------
__global__ void elu_kernel(const float* __restrict__ in, float* __restrict__ out,
                           int C, int ldin, long total) {
    long p = (long)blockIdx.x * blockDim.x + threadIdx.x;
    if (p >= total) return;
    long r = p / C; int c = (int)(p % C);
    float x = in[r * ldin + c];
    out[p] = (x > 0.f) ? x : (__expf(x) - 1.f);
}

// ---------------------------------------------------------------------------
// Attention fusion over the two views (per node):
// v = tanh(emb @ w_omega); vu = v @ u_omega; alpha = softmax(vu + 1e-6)
// fused = alpha0*e1 + alpha1*e2
// ---------------------------------------------------------------------------
__global__ __launch_bounds__(128) void fuse_kernel(
        const float* __restrict__ e1, const float* __restrict__ e2,
        const float* __restrict__ w_omega, const float* __restrict__ u_omega,
        float* __restrict__ fused, float* __restrict__ alpha_out) {
    __shared__ float sE[2][64];
    __shared__ float sV[128];
    __shared__ float sA[2];
    int n = blockIdx.x, t = threadIdx.x;
    int k = t >> 6, d = t & 63;
    sE[k][d] = (k == 0 ? e1 : e2)[(size_t)n * 64 + d];
    __syncthreads();
    float acc = 0.f;
#pragma unroll 8
    for (int j = 0; j < 64; ++j) acc += sE[k][j] * w_omega[j * 64 + d];
    sV[t] = tanhf(acc) * u_omega[d];
    __syncthreads();
    for (int s = 32; s > 0; s >>= 1) {
        if (d < s) sV[t] += sV[t + s];
        __syncthreads();
    }
    if (t == 0) {
        float vu0 = sV[0] + 1e-6f, vu1 = sV[64] + 1e-6f;
        float M = fmaxf(vu0, vu1);
        float x0 = __expf(vu0 - M), x1 = __expf(vu1 - M);
        float inv = 1.0f / (x0 + x1);
        sA[0] = x0 * inv; sA[1] = x1 * inv;
        alpha_out[(size_t)n * 2 + 0] = sA[0];
        alpha_out[(size_t)n * 2 + 1] = sA[1];
    }
    __syncthreads();
    if (k == 0)
        fused[(size_t)n * 64 + d] = sA[0] * sE[0][d] + sA[1] * sE[1][d];
}

// ---------------------------------------------------------------------------
// Host-side orchestration
// ---------------------------------------------------------------------------
namespace {

constexpr int N_NODES = 4096;

struct Scratch {
    unsigned short* hA;    // [4096, 3008] bf16
    unsigned short* Wt;    // [3072, 3008] bf16
    unsigned short* WhT;   // [3072, 4096] bf16
    unsigned short* att;   // [4096, 4096] bf16
    float* mid;            // [4096, 3072] f32  (Wh, then att@Wh)
    float* s1;  float* s2; // [4096]
    float* h_hid;          // [4096, 512]
    float* feat_emb; float* sp_emb; float* corr_f; float* corr_s;  // [4096,64]
    float* alpha_scratch;  // [4096,2]
};

inline int ceil_to(int x, int n) { return (x + n - 1) / n * n; }

void run_gat(hipStream_t st, const float* h, const int* adj,
             const float* W, const float* a, float* out,
             int Fin, int Fout, const Scratch& w) {
    const int N  = N_NODES;
    const int KP = ceil_to(Fin, 64);    // K must be a multiple of 64 (pipeline)
    const int NP = ceil_to(Fout, 256);
    // 1) h (f32, [N,Fin]) -> hA bf16 [N,KP]
    {
        long total = (long)N * KP;
        cast_pad_kernel<<<(total + 255) / 256, 256, 0, st>>>(h, w.hA, Fin, KP, total);
    }
    // 2) W [Fin,Fout] -> Wt bf16 [NP,KP] (B^T, zero padded)
    transpose_cast_kernel<<<dim3(KP / 16, NP / 16), 256, 0, st>>>(
        W, w.Wt, Fin, Fout, Fout, NP, KP);
    // 3) Wh = h @ W  (WMMA)
    gemm_bf16_wmma_kernel<<<dim3(NP / 256, N / 32), 128, 0, st>>>(w.hA, w.Wt, w.mid, KP, NP);
    // 4) row scores
    s1s2_kernel<<<N, 256, 0, st>>>(w.mid, a, w.s1, w.s2, Fout, NP);
    // 5) masked softmax -> att (bf16)
    attn_softmax_kernel<<<N, 256, 0, st>>>(adj, w.s1, w.s2, w.att, N);
    // 6) Wh [N,Fout] -> WhT bf16 [NP, N]
    transpose_cast_kernel<<<dim3(N / 16, NP / 16), 256, 0, st>>>(
        w.mid, w.WhT, N, Fout, NP, NP, N);
    // 7) Z = att @ Wh  (WMMA, K = N = 4096)
    gemm_bf16_wmma_kernel<<<dim3(NP / 256, N / 32), 128, 0, st>>>(w.att, w.WhT, w.mid, N, NP);
    // 8) out = elu(Z)  ([N,Fout] contiguous)
    {
        long total = (long)N * Fout;
        elu_kernel<<<(total + 255) / 256, 256, 0, st>>>(w.mid, out, Fout, NP, total);
    }
}

} // namespace

extern "C" void kernel_launch(void* const* d_in, const int* in_sizes, int n_in,
                              void* d_out, int out_size, void* d_ws, size_t ws_size,
                              hipStream_t stream) {
    (void)in_sizes; (void)n_in; (void)out_size; (void)ws_size;

    const float* features    = (const float*)d_in[0];   // [4096,3000]
    const int*   feat_adj    = (const int*)  d_in[1];   // [4096,4096]
    const int*   spatial_adj = (const int*)  d_in[2];
    const float* W_enc1 = (const float*)d_in[3];
    const float* a_enc1 = (const float*)d_in[4];
    const float* W_enc2 = (const float*)d_in[5];
    const float* a_enc2 = (const float*)d_in[6];
    const float* W_dec1 = (const float*)d_in[7];
    const float* a_dec1 = (const float*)d_in[8];
    const float* W_dec2 = (const float*)d_in[9];
    const float* a_dec2 = (const float*)d_in[10];
    const float* w_omega = (const float*)d_in[11];
    const float* u_omega = (const float*)d_in[12];

    // d_out layout: emb_latent[4096*64] | recon[4096*3000] | corr[4096*64] | alpha[4096*2]
    float* out_emb   = (float*)d_out;
    float* out_recon = out_emb + (size_t)N_NODES * 64;
    float* out_corr  = out_recon + (size_t)N_NODES * 3000;
    float* out_alpha = out_corr + (size_t)N_NODES * 64;

    // Workspace carve-up (256B aligned slices)
    char* p = (char*)d_ws;
    auto take = [&](size_t bytes) -> void* {
        void* r = (void*)p;
        p += (bytes + 255) & ~(size_t)255;
        return r;
    };
    Scratch w;
    w.hA  = (unsigned short*)take((size_t)4096 * 3008 * 2);
    w.Wt  = (unsigned short*)take((size_t)3072 * 3008 * 2);
    w.WhT = (unsigned short*)take((size_t)3072 * 4096 * 2);
    w.att = (unsigned short*)take((size_t)4096 * 4096 * 2);
    w.mid = (float*)take((size_t)4096 * 3072 * 4);
    w.s1  = (float*)take((size_t)4096 * 4);
    w.s2  = (float*)take((size_t)4096 * 4);
    w.h_hid    = (float*)take((size_t)4096 * 512 * 4);
    w.feat_emb = (float*)take((size_t)4096 * 64 * 4);
    w.sp_emb   = (float*)take((size_t)4096 * 64 * 4);
    w.corr_f   = (float*)take((size_t)4096 * 64 * 4);
    w.corr_s   = (float*)take((size_t)4096 * 64 * 4);
    w.alpha_scratch = (float*)take((size_t)4096 * 2 * 4);

    // --- encoder on the feature graph ---
    run_gat(stream, features, feat_adj, W_enc1, a_enc1, w.h_hid,   3000, 512, w);
    run_gat(stream, w.h_hid,  feat_adj, W_enc2, a_enc2, w.feat_emb, 512,  64, w);
    // --- encoder on the spatial graph ---
    run_gat(stream, features, spatial_adj, W_enc1, a_enc1, w.h_hid,  3000, 512, w);
    run_gat(stream, w.h_hid,  spatial_adj, W_enc2, a_enc2, w.sp_emb,  512,  64, w);
    // --- fuse views -> emb_latent, alpha ---
    fuse_kernel<<<N_NODES, 128, 0, stream>>>(w.feat_emb, w.sp_emb, w_omega, u_omega,
                                             out_emb, out_alpha);
    // --- decoder (spatial graph) -> recon ---
    run_gat(stream, out_emb, spatial_adj, W_dec1, a_dec1, w.h_hid,   64, 512, w);
    run_gat(stream, w.h_hid, spatial_adj, W_dec2, a_dec2, out_recon, 512, 3000, w);
    // --- re-encode recon on both graphs ---
    run_gat(stream, out_recon, feat_adj, W_enc1, a_enc1, w.h_hid,  3000, 512, w);
    run_gat(stream, w.h_hid,   feat_adj, W_enc2, a_enc2, w.corr_f,  512,  64, w);
    run_gat(stream, out_recon, spatial_adj, W_enc1, a_enc1, w.h_hid, 3000, 512, w);
    run_gat(stream, w.h_hid,   spatial_adj, W_enc2, a_enc2, w.corr_s, 512,  64, w);
    // --- fuse corr views (alpha discarded per reference) ---
    fuse_kernel<<<N_NODES, 128, 0, stream>>>(w.corr_f, w.corr_s, w_omega, u_omega,
                                             out_corr, w.alpha_scratch);
}
